// MultiHead_88862873354891
// MI455X (gfx1250) — compile-verified, matrix operand order checked by reading, not run on hardware
//
#include <hip/hip_runtime.h>
#include <hip/hip_bf16.h>

// ---------------------------------------------------------------------------
// MultiHead attention for MI455X (gfx1250), bf16 WMMA everywhere.
// B=4, S=2048, D=1024, H=16, dk=64.
// ---------------------------------------------------------------------------

#define S_LEN 2048
#define D_MODEL 1024
#define NHEADS 16
#define DK 64

typedef __attribute__((ext_vector_type(8)))  float  v8f;
typedef __attribute__((ext_vector_type(8)))  __bf16 v8bf;
typedef __attribute__((ext_vector_type(16))) __bf16 v16bf;

union BFrag { v16bf v; v8bf h[2]; };

// round-to-nearest-even f32 -> bf16 bits (scalar)
__device__ __forceinline__ unsigned short f2bf(float f) {
    unsigned u = __builtin_bit_cast(unsigned, f);
    u += 0x7FFFu + ((u >> 16) & 1u);
    return (unsigned short)(u >> 16);
}

// pack two f32 into {lo,hi} bf16 pair in one u32 (single VOP3 on gfx1250)
__device__ __forceinline__ unsigned pk2bf(float lo, float hi) {
#if __has_builtin(__builtin_amdgcn_cvt_pk_bf16_f32)
    typedef __attribute__((ext_vector_type(2))) __bf16 v2bf;
    v2bf r = __builtin_amdgcn_cvt_pk_bf16_f32(lo, hi);
    return __builtin_bit_cast(unsigned, r);
#else
    unsigned r;
    asm("v_cvt_pk_bf16_f32 %0, %1, %2" : "=v"(r) : "v"(lo), "v"(hi));
    return r;
#endif
}

__device__ __forceinline__ unsigned pk2u16(unsigned short lo, unsigned short hi) {
    return (unsigned)lo | ((unsigned)hi << 16);
}

// raw v_exp_f32 (exp2); -inf -> 0, underflow flushes — fine for softmax
__device__ __forceinline__ float fexp2(float x) {
#if __has_builtin(__builtin_amdgcn_exp2f)
    return __builtin_amdgcn_exp2f(x);
#else
    return exp2f(x);
#endif
}

// Load one WMMA 16x32 bf16 fragment chunk for this lane.
// base = row (lane%16) start; khalf = lane>>4. Per ISA layout:
// elems 0..7 = K[8*khalf..+7], elems 8..15 = K[16+8*khalf..+7].
__device__ __forceinline__ v16bf load_frag(const unsigned short* base, int khalf) {
    BFrag f;
    f.h[0] = *(const v8bf*)(base + khalf * 8);
    f.h[1] = *(const v8bf*)(base + 16 + khalf * 8);
    return f.v;
}

__device__ __forceinline__ v8f wmma_bf16(v16bf a, v16bf b, v8f c) {
    return __builtin_amdgcn_wmma_f32_16x16x32_bf16(
        false, a, false, b, (short)0, c, false, false);
}

// ----- CDNA5 async global->LDS copy (16B per lane per call) ----------------
#if __has_builtin(__builtin_amdgcn_global_load_async_to_lds_b128)
#define HAVE_ASYNC_LDS 1
typedef __attribute__((__vector_size__(4 * sizeof(int)))) int v4i32;
typedef __attribute__((address_space(1))) v4i32 gv4i32;
typedef __attribute__((address_space(3))) v4i32 sv4i32;
__device__ __forceinline__ void cp16_async(unsigned short* lds_dst,
                                           const unsigned short* gsrc) {
    __builtin_amdgcn_global_load_async_to_lds_b128(
        (gv4i32*)(unsigned long long)(size_t)gsrc,
        (sv4i32*)(unsigned)(size_t)lds_dst, 0, 0);
}
__device__ __forceinline__ void async_wait0() {
    asm volatile("s_wait_asynccnt 0" ::: "memory");
}
#endif

// 32B bf16 copy global->LDS: async when available, else vector copy.
__device__ __forceinline__ void cp32(unsigned short* lds_dst,
                                     const unsigned short* gsrc) {
#if defined(HAVE_ASYNC_LDS)
    cp16_async(lds_dst, gsrc);
    cp16_async(lds_dst + 8, gsrc + 8);
#else
    ((uint4*)lds_dst)[0] = ((const uint4*)gsrc)[0];
    ((uint4*)lds_dst)[1] = ((const uint4*)gsrc)[1];
#endif
}

__device__ __forceinline__ void stage_wait() {
#if defined(HAVE_ASYNC_LDS)
    async_wait0();
#endif
    __syncthreads();
}

// ---------------------------------------------------------------------------
// Kernel 0: weight prep  Wt_bf16[n][k] = cvt(W_f32[k][n])  (run once per W)
// ---------------------------------------------------------------------------
__global__ __launch_bounds__(256) void k_wprep(const float* __restrict__ W,
                                               unsigned short* __restrict__ Wt) {
    __shared__ unsigned short sT[64 * 64];   // [n][k]
    const int tid = threadIdx.x;
    const int k0 = blockIdx.x * 64, n0 = blockIdx.y * 64;
    {
        int kp = (tid & 31) * 2, nb = (tid >> 5) * 8;
        const float* s0 = W + (size_t)(k0 + kp) * D_MODEL + n0 + nb;
        const float* s1 = s0 + D_MODEL;
#pragma unroll
        for (int i = 0; i < 8; ++i)
            ((unsigned*)sT)[((nb + i) * 64 + kp) >> 1] = pk2bf(s0[i], s1[i]);
    }
    __syncthreads();
    {
        int nn = tid >> 2, kb = (tid & 3) * 16;
        const unsigned short* src = sT + nn * 64 + kb;
        unsigned short* dst = Wt + (size_t)(n0 + nn) * D_MODEL + k0 + kb;
        ((uint4*)dst)[0] = ((const uint4*)src)[0];
        ((uint4*)dst)[1] = ((const uint4*)src)[1];
    }
}

// ---------------------------------------------------------------------------
// Kernel 1: projection GEMM  Out_headmajor = X[8192x1024] * W
// W pre-transposed bf16 [n][k]. Output bf16 [B][H][S][DK].
// Block 256 threads = 8 waves; tile 128x128; wave 32x64 (2x4 frags); K=32.
// ---------------------------------------------------------------------------
__global__ __launch_bounds__(256) void k_proj(const float* __restrict__ X,
                                              const unsigned short* __restrict__ Wt,
                                              unsigned short* __restrict__ Out) {
    __shared__ unsigned short sA[128 * 32];   // [m][k]
    __shared__ unsigned short sB[128 * 32];   // [n][k]

    const int tid  = threadIdx.x;
    const int lane = tid & 31;
    const int wave = tid >> 5;
    const int m0 = blockIdx.x * 128;
    const int n0 = blockIdx.y * 128;
    const int wr = (wave >> 1) * 32;
    const int wc = (wave & 1) * 64;
    const int r = lane & 15, khalf = lane >> 4;
    const int srow = tid >> 1, scb = (tid & 1) * 16;

    v8f c[2][4] = {{{}, {}, {}, {}}, {{}, {}, {}, {}}};

    for (int k0 = 0; k0 < D_MODEL; k0 += 32) {
        // A: 128x32 f32 -> bf16 LDS (packed cvt, u32 stores). 2 threads/row.
        {
            const float* src = X + (size_t)(m0 + srow) * D_MODEL + k0 + scb;
            unsigned* dst = (unsigned*)sA + (srow * 32 + scb) / 2;
#pragma unroll
            for (int i = 0; i < 8; ++i) dst[i] = pk2bf(src[2 * i], src[2 * i + 1]);
            if (k0 + 32 < D_MODEL) __builtin_prefetch(src + 32, 0, 1);
        }
        // B: pure bf16 copy of Wt rows (async global->LDS).
        {
            const unsigned short* src = Wt + (size_t)(n0 + srow) * D_MODEL + k0 + scb;
            cp32(sB + srow * 32 + scb, src);
            if (k0 + 32 < D_MODEL) __builtin_prefetch(src + 32, 0, 1);
        }
        stage_wait();

        v16bf af[2], bf_[4];
#pragma unroll
        for (int i = 0; i < 2; ++i)
            af[i] = load_frag(sA + (wr + i * 16 + r) * 32, khalf);
#pragma unroll
        for (int j = 0; j < 4; ++j)
            bf_[j] = load_frag(sB + (wc + j * 16 + r) * 32, khalf);
#pragma unroll
        for (int i = 0; i < 2; ++i)
#pragma unroll
            for (int j = 0; j < 4; ++j)
                c[i][j] = wmma_bf16(af[i], bf_[j], c[i][j]);
        __syncthreads();
    }

    // Epilogue: C layout -> head-major bf16 [B][H][S][DK]
    const int col = lane & 15, hs = lane >> 4;
#pragma unroll
    for (int fi = 0; fi < 2; ++fi)
#pragma unroll
        for (int fj = 0; fj < 4; ++fj) {
#pragma unroll
            for (int j = 0; j < 8; ++j) {
                int m = m0 + wr + fi * 16 + j + 8 * hs;
                int n = n0 + wc + fj * 16 + col;
                int b = m >> 11, s = m & (S_LEN - 1);
                int h = n >> 6, d = n & (DK - 1);
                Out[(((size_t)(b * NHEADS + h)) * S_LEN + s) * DK + d] =
                    f2bf(c[fi][fj][j]);
            }
        }
}

// ---------------------------------------------------------------------------
// Kernel 2: causal flash attention. Grid (S/64, B*H), 128 threads = 4 waves,
// each wave owns 16 query rows. Key tiles of 64 staged in LDS
// (K row-major via async copy; V transposed for PV WMMA).
// Softmax in exp2 domain; row-sum kept as per-lane partials (one final
// 16-lane reduction) since the rescale factor is lane-uniform per row.
// ---------------------------------------------------------------------------
__global__ __launch_bounds__(128) void k_attn(const unsigned short* __restrict__ Qh,
                                              const unsigned short* __restrict__ Kh,
                                              const unsigned short* __restrict__ Vh,
                                              unsigned short* __restrict__ AO) {
    __shared__ unsigned short sK[64 * 64];      // [t][d]
    __shared__ unsigned short sVt[64 * 64];     // [d][t]
    __shared__ unsigned short sP[4][16 * 64];   // per-wave P tile [q][t]

    const int tid  = threadIdx.x;
    const int lane = tid & 31;
    const int wave = tid >> 5;
    const int bh = blockIdx.y;
    const int q0 = blockIdx.x * 64;
    const int wq = q0 + wave * 16;

    const unsigned short* Qb = Qh + (size_t)bh * S_LEN * DK;
    const unsigned short* Kb = Kh + (size_t)bh * S_LEN * DK;
    const unsigned short* Vb = Vh + (size_t)bh * S_LEN * DK;

    const int r = lane & 15, khalf = lane >> 4;
    const int col = r, hs = khalf;

    // Q fragments for this wave's 16 rows (d 0..31 and 32..63), kept live.
    v16bf qa0 = load_frag(Qb + (size_t)(wq + r) * DK, khalf);
    v16bf qa1 = load_frag(Qb + (size_t)(wq + r) * DK + 32, khalf);

    v8f acc[4] = {{}, {}, {}, {}};
    float mrow[8], lrow[8];   // lrow: per-lane partial sums
#pragma unroll
    for (int j = 0; j < 8; ++j) { mrow[j] = -INFINITY; lrow[j] = 0.0f; }

    // 1/sqrt(64) * log2(e): softmax runs in the exp2 domain.
    const float scale2 = 0.125f * 1.44269504088896340736f;
    const int ntiles = blockIdx.x + 1;

    for (int kt = 0; kt < ntiles; ++kt) {
        const int t0 = kt * 64;
        // --- stage K tile [64 keys][64 d] (pure bf16 copy: 32B/thread) ----
        {
            int key = tid & 63, half = tid >> 6;
            const unsigned short* ks = Kb + (size_t)(t0 + key) * DK + half * 32;
            cp32(sK + key * 64 + half * 32, ks);
        }
        // --- stage V tile transposed sVt[d][t], k-pairs packed ------------
        {
            int db = (tid & 3) * 16, kp = (tid >> 2) * 2;
            union { uint4 q[2]; unsigned short u[16]; } va, vb2;
            const unsigned short* v0 = Vb + (size_t)(t0 + kp) * DK + db;
            const unsigned short* v1 = v0 + DK;
            va.q[0]  = ((const uint4*)v0)[0]; va.q[1]  = ((const uint4*)v0)[1];
            vb2.q[0] = ((const uint4*)v1)[0]; vb2.q[1] = ((const uint4*)v1)[1];
#pragma unroll
            for (int i = 0; i < 16; ++i)
                ((unsigned*)sVt)[((db + i) * 64 + kp) >> 1] =
                    pk2u16(va.u[i], vb2.u[i]);
        }
        stage_wait();

        // --- scores: S = Q K^T over d=64 (2 WMMA per 16-key group) --------
        v8f scf[4];
#pragma unroll
        for (int g = 0; g < 4; ++g) {
            v16bf kba = load_frag(sK + (g * 16 + r) * 64, khalf);
            v16bf kbb = load_frag(sK + (g * 16 + r) * 64 + 32, khalf);
            v8f s = {};
            s = wmma_bf16(qa0, kba, s);
            s = wmma_bf16(qa1, kbb, s);
            scf[g] = s;
        }

        // --- online softmax. C layout: row = j + 8*(lane/16), col = lane%16
        float rescale[8];
        v8f pe[4] = {{}, {}, {}, {}};
#pragma unroll
        for (int j = 0; j < 8; ++j) {
            int qrow = wq + j + 8 * hs;
            float vv[4];
            float mx = -INFINITY;
#pragma unroll
            for (int g = 0; g < 4; ++g) {
                int key = t0 + g * 16 + col;
                vv[g] = (key <= qrow) ? scf[g][j] * scale2 : -INFINITY;
                mx = fmaxf(mx, vv[g]);
            }
#pragma unroll
            for (int off = 1; off < 16; off <<= 1)
                mx = fmaxf(mx, __shfl_xor(mx, off, 32));
            float nm = fmaxf(mrow[j], mx);          // lane-uniform per row
            float sc = fexp2(mrow[j] - nm);
            float rs = 0.0f;
#pragma unroll
            for (int g = 0; g < 4; ++g) {
                float e = fexp2(vv[g] - nm);
                pe[g][j] = e;
                rs += e;
            }
            lrow[j] = lrow[j] * sc + rs;            // per-lane partial
            mrow[j] = nm;
            rescale[j] = sc;
        }
#pragma unroll
        for (int f = 0; f < 4; ++f)
#pragma unroll
            for (int j = 0; j < 8; ++j) acc[f][j] *= rescale[j];

        // --- P (C layout) -> wave-private LDS -> A fragments --------------
        unsigned short* Pw = sP[wave];
#pragma unroll
        for (int j = 0; j < 8; ++j) {
            int row = j + 8 * hs;
#pragma unroll
            for (int g = 0; g < 4; ++g)
                Pw[row * 64 + g * 16 + col] = f2bf(pe[g][j]);
        }
        v16bf pa0 = load_frag(Pw + r * 64, khalf);
        v16bf pa1 = load_frag(Pw + r * 64 + 32, khalf);

        // --- out += P * V (B from transposed V) ---------------------------
#pragma unroll
        for (int f = 0; f < 4; ++f) {
            v16bf vb0 = load_frag(sVt + (f * 16 + r) * 64, khalf);
            acc[f] = wmma_bf16(pa0, vb0, acc[f]);
            v16bf vb1 = load_frag(sVt + (f * 16 + r) * 64 + 32, khalf);
            acc[f] = wmma_bf16(pa1, vb1, acc[f]);
        }
        __syncthreads();
    }

    // Final row-sum reduction across the 16 columns, then reciprocal (once).
    float rinv[8];
#pragma unroll
    for (int j = 0; j < 8; ++j) {
#pragma unroll
        for (int off = 1; off < 16; off <<= 1)
            lrow[j] += __shfl_xor(lrow[j], off, 32);
        rinv[j] = 1.0f / lrow[j];
    }

    // Normalize, store bf16 to [B][S][H*64+d] (row-major for final GEMM).
    const int b = bh >> 4, h = bh & (NHEADS - 1);
#pragma unroll
    for (int f = 0; f < 4; ++f)
#pragma unroll
        for (int j = 0; j < 8; ++j) {
            int qrow = wq + j + 8 * hs;
            int d = f * 16 + col;
            float o = acc[f][j] * rinv[j];
            AO[((size_t)b * S_LEN + qrow) * D_MODEL + h * DK + d] = f2bf(o);
        }
}

// ---------------------------------------------------------------------------
// Kernel 3: output projection  out_f32 = AO_bf16[8192x1024] * Wot_bf16
// Both operands bf16 -> staging is pure async copies.
// ---------------------------------------------------------------------------
__global__ __launch_bounds__(256) void k_oproj(const unsigned short* __restrict__ A,
                                               const unsigned short* __restrict__ Wt,
                                               float* __restrict__ Out) {
    __shared__ unsigned short sA[128 * 32];
    __shared__ unsigned short sB[128 * 32];

    const int tid  = threadIdx.x;
    const int lane = tid & 31;
    const int wave = tid >> 5;
    const int m0 = blockIdx.x * 128;
    const int n0 = blockIdx.y * 128;
    const int wr = (wave >> 1) * 32;
    const int wc = (wave & 1) * 64;
    const int r = lane & 15, khalf = lane >> 4;
    const int srow = tid >> 1, scb = (tid & 1) * 16;

    v8f c[2][4] = {{{}, {}, {}, {}}, {{}, {}, {}, {}}};

    for (int k0 = 0; k0 < D_MODEL; k0 += 32) {
        {
            const unsigned short* src = A + (size_t)(m0 + srow) * D_MODEL + k0 + scb;
            cp32(sA + srow * 32 + scb, src);
            if (k0 + 32 < D_MODEL) __builtin_prefetch(src + 32, 0, 1);
        }
        {
            const unsigned short* src = Wt + (size_t)(n0 + srow) * D_MODEL + k0 + scb;
            cp32(sB + srow * 32 + scb, src);
            if (k0 + 32 < D_MODEL) __builtin_prefetch(src + 32, 0, 1);
        }
        stage_wait();

        v16bf af[2], bf_[4];
#pragma unroll
        for (int i = 0; i < 2; ++i)
            af[i] = load_frag(sA + (wr + i * 16 + r) * 32, khalf);
#pragma unroll
        for (int j = 0; j < 4; ++j)
            bf_[j] = load_frag(sB + (wc + j * 16 + r) * 32, khalf);
#pragma unroll
        for (int i = 0; i < 2; ++i)
#pragma unroll
            for (int j = 0; j < 4; ++j)
                c[i][j] = wmma_bf16(af[i], bf_[j], c[i][j]);
        __syncthreads();
    }

    const int col = lane & 15, hs = lane >> 4;
#pragma unroll
    for (int fi = 0; fi < 2; ++fi)
#pragma unroll
        for (int fj = 0; fj < 4; ++fj) {
#pragma unroll
            for (int j = 0; j < 8; ++j) {
                int m = m0 + wr + fi * 16 + j + 8 * hs;
                int n = n0 + wc + fj * 16 + col;
                Out[(size_t)m * D_MODEL + n] = c[fi][fj][j];
            }
        }
}

// ---------------------------------------------------------------------------
// Host-side launcher. Workspace (bf16 elems):
//   Qh | Kh | Vh | AO (8Mi each) | Wqt | Wkt | Wvt | Wot (1Mi each) = 72 MB.
// ---------------------------------------------------------------------------
extern "C" void kernel_launch(void* const* d_in, const int* in_sizes, int n_in,
                              void* d_out, int out_size, void* d_ws, size_t ws_size,
                              hipStream_t stream) {
    const float* q  = (const float*)d_in[0];
    const float* k  = (const float*)d_in[1];
    const float* v  = (const float*)d_in[2];
    const float* wq = (const float*)d_in[3];
    const float* wk = (const float*)d_in[4];
    const float* wv = (const float*)d_in[5];
    const float* wo = (const float*)d_in[6];

    const size_t headElems = (size_t)4 * NHEADS * S_LEN * DK;  // 8Mi
    const size_t wElems = (size_t)D_MODEL * D_MODEL;           // 1Mi
    unsigned short* Qh  = (unsigned short*)d_ws;
    unsigned short* Kh  = Qh + headElems;
    unsigned short* Vh  = Kh + headElems;
    unsigned short* AO  = Vh + headElems;
    unsigned short* Wqt = AO + headElems;
    unsigned short* Wkt = Wqt + wElems;
    unsigned short* Wvt = Wkt + wElems;
    unsigned short* Wot = Wvt + wElems;

    dim3 prepGrid(16, 16);
    k_wprep<<<prepGrid, 256, 0, stream>>>(wq, Wqt);
    k_wprep<<<prepGrid, 256, 0, stream>>>(wk, Wkt);
    k_wprep<<<prepGrid, 256, 0, stream>>>(wv, Wvt);
    k_wprep<<<prepGrid, 256, 0, stream>>>(wo, Wot);

    dim3 gemmGrid(64, 8);     // M/128, N/128
    dim3 gemmBlk(256);
    k_proj<<<gemmGrid, gemmBlk, 0, stream>>>(q, Wqt, Qh);
    k_proj<<<gemmGrid, gemmBlk, 0, stream>>>(k, Wkt, Kh);
    k_proj<<<gemmGrid, gemmBlk, 0, stream>>>(v, Wvt, Vh);

    dim3 attnGrid(S_LEN / 64, 4 * NHEADS);  // (32, 64)
    k_attn<<<attnGrid, dim3(128), 0, stream>>>(Qh, Kh, Vh, AO);

    k_oproj<<<gemmGrid, gemmBlk, 0, stream>>>(AO, Wot, (float*)d_out);
}
